// GAT6_56478819943011
// MI455X (gfx1250) — matrix operand order repaired and checked by reading, not compiled, not run on hardware
//
#include <hip/hip_runtime.h>
#include <hip/hip_bf16.h>

// ---------------------------------------------------------------------------
// GATv2 x6 + global max pool + MLP head for MI455X (gfx1250, wave32, WMMA)
// ---------------------------------------------------------------------------

typedef __attribute__((ext_vector_type(16))) _Float16 v16h;
typedef __attribute__((ext_vector_type(8)))  _Float16 v8h;
typedef __attribute__((ext_vector_type(8)))  float    v8f;

#define N_NODES 20000
#define N_EDGES 320000
#define N_GRAPHS 16

// ---------------------------------------------------------------------------
// Repack W[K,Fo] (f32, row-major) into f16 WMMA B-fragment order:
//   Bf[((nt*(K/32)+ks)*32 + lane)*16 + elem]
// where per the CDNA5 ISA 16-bit B layout (32x16 tile, 8 VGPRs):
//   k = ks*32 + (elem&7) + 8*(lane>>4) + 16*(elem>>3),  n = nt*16 + (lane&15)
// ---------------------------------------------------------------------------
__global__ void repack_w_kernel(const float* __restrict__ W,
                                _Float16* __restrict__ Bf, int K, int Fo) {
    int idx = blockIdx.x * blockDim.x + threadIdx.x;
    int total = K * Fo;
    if (idx >= total) return;
    int elem = idx & 15;
    int lane = (idx >> 4) & 31;
    int frag = idx >> 9;               // nt*(K/32) + ks
    int ksteps = K >> 5;
    int nt = frag / ksteps;
    int ks = frag - nt * ksteps;
    int k = ks * 32 + (elem & 7) + ((lane >> 4) << 3) + ((elem >> 3) << 4);
    int n = nt * 16 + (lane & 15);
    Bf[idx] = (_Float16)W[(size_t)k * Fo + n];
}

// ---------------------------------------------------------------------------
// C[M,Fo] (f32) = A[M,K] (f16 row-major) @ Bfrag   via v_wmma_f32_16x16x32_f16
// One wave per (16-row tile) x (NT N-tiles). NT is a compile-time constant so
// the inner loop is branch-free straight-line code (no EXEC games, no register
// rotation) and EXEC is trivially all-1s at every WMMA.
// A-fragment per CDNA5 16-bit A layout: lane row = lane&15; per-lane halves
// are two contiguous 8-half runs at k = kb + 8*(lane>>4) and +16.
// ---------------------------------------------------------------------------
template <int NT>
__global__ void __launch_bounds__(256)
gemm_wmma_kernel(const _Float16* __restrict__ A, const _Float16* __restrict__ Bf,
                 float* __restrict__ C, int M, int K, int Fo) {
    int wave = blockIdx.x * (blockDim.x >> 5) + (threadIdx.x >> 5);
    int lane = threadIdx.x & 31;
    int mtiles = M >> 4;
    int ntiles = Fo >> 4;
    int ngroups = ntiles / NT;                 // exact by construction
    if (wave >= mtiles * ngroups) return;      // wave-uniform guard

    int mt = wave % mtiles;
    int ng = wave / mtiles;
    int nt0 = ng * NT;

    int row  = (mt << 4) + (lane & 15);
    int kb8  = (lane >> 4) << 3;               // 0 or 8
    const _Float16* arow = A + (size_t)row * K;

    int ksteps = K >> 5;
    const _Float16* bbase = Bf + (((size_t)nt0 * ksteps * 32 + lane) << 4);
    const size_t bstride_nt = (size_t)ksteps * 32 * 16;   // halves per N-tile
    const size_t bstride_ks = (size_t)32 * 16;            // halves per K-step

    v8f acc[NT];
#pragma unroll
    for (int t = 0; t < NT; ++t) acc[t] = (v8f){0.f,0.f,0.f,0.f,0.f,0.f,0.f,0.f};

    for (int ks = 0; ks < ksteps; ++ks) {
        int kb = ks << 5;
        union { v16h v; v8h h[2]; } af;
        af.h[0] = *(const v8h*)(arow + kb + kb8);        // elems 0..7
        af.h[1] = *(const v8h*)(arow + kb + kb8 + 16);   // elems 8..15
        const _Float16* bks = bbase + (size_t)ks * bstride_ks;
#pragma unroll
        for (int t = 0; t < NT; ++t) {
            v16h bfrag = *(const v16h*)(bks + (size_t)t * bstride_nt);
            acc[t] = __builtin_amdgcn_wmma_f32_16x16x32_f16(
                false, af.v, false, bfrag, (short)0, acc[t], false, false);
        }
    }

    // C/D layout: VGPR j -> row = j + 8*(lane>>4), col = lane&15
    int crow0 = (mt << 4) + ((lane >> 4) << 3);
#pragma unroll
    for (int t = 0; t < NT; ++t) {
        int col = ((nt0 + t) << 4) + (lane & 15);
#pragma unroll
        for (int j = 0; j < 8; ++j)
            C[(size_t)(crow0 + j) * Fo + col] = acc[t][j];
    }
}

// ---------------------------------------------------------------------------
// Softmax state init and bias seeding
// ---------------------------------------------------------------------------
__global__ void init_softmax_kernel(float* __restrict__ m, float* __restrict__ denom, int n) {
    int i = blockIdx.x * blockDim.x + threadIdx.x;
    if (i < n) { m[i] = -1e30f; denom[i] = 0.f; }
}

__global__ void init_bias_rows_kernel(float* __restrict__ out, const float* __restrict__ b,
                                      int n, int F) {
    int i = blockIdx.x * blockDim.x + threadIdx.x;
    if (i < n * F) out[i] = b[i % F];
}

// ---------------------------------------------------------------------------
// Edge pass 1: e = att . LeakyReLU(xl[src] + xr[dst], 0.2); segment max via
// float atomicMax (-> global_atomic_max_num_f32). One wave per edge.
// ---------------------------------------------------------------------------
__global__ void edge_scores_kernel(const float* __restrict__ xl, const float* __restrict__ xr,
                                   const int* __restrict__ src, const int* __restrict__ dst,
                                   const float* __restrict__ att,
                                   float* __restrict__ evals, float* __restrict__ m,
                                   int E, int Fo) {
    int e    = blockIdx.x * (blockDim.x >> 5) + (threadIdx.x >> 5);
    int lane = threadIdx.x & 31;
    if (e >= E) return;
    int s = src[e], d = dst[e];
    const float* pl = xl + (size_t)s * Fo;
    const float* pr = xr + (size_t)d * Fo;
    float acc = 0.f;
    for (int f = lane; f < Fo; f += 32) {
        float v = pl[f] + pr[f];
        v = (v > 0.f) ? v : 0.2f * v;
        acc += v * att[f];
    }
#pragma unroll
    for (int o = 16; o > 0; o >>= 1) acc += __shfl_xor(acc, o, 32);
    if (lane == 0) {
        evals[e] = acc;
        atomicMax(m + d, acc);
    }
}

// ---------------------------------------------------------------------------
// Edge pass 2: ex = exp(e - m[dst]); denom[dst] += ex
// ---------------------------------------------------------------------------
__global__ void edge_exp_kernel(const float* __restrict__ evals, const int* __restrict__ dst,
                                const float* __restrict__ m, float* __restrict__ ex,
                                float* __restrict__ denom, int E) {
    int e = blockIdx.x * blockDim.x + threadIdx.x;
    if (e >= E) return;
    int d = dst[e];
    float v = expf(evals[e] - m[d]);
    ex[e] = v;
    atomicAdd(denom + d, v);
}

// ---------------------------------------------------------------------------
// Edge pass 3: out[dst] += (ex/(denom[dst]+1e-16)) * xl[src]. One wave/edge.
// ---------------------------------------------------------------------------
__global__ void edge_aggregate_kernel(const float* __restrict__ xl, const float* __restrict__ ex,
                                      const float* __restrict__ denom,
                                      const int* __restrict__ src, const int* __restrict__ dst,
                                      float* __restrict__ out, int E, int Fo) {
    int e    = blockIdx.x * (blockDim.x >> 5) + (threadIdx.x >> 5);
    int lane = threadIdx.x & 31;
    if (e >= E) return;
    int s = src[e], d = dst[e];
    float alpha = ex[e] / (denom[d] + 1e-16f);
    const float* pl = xl + (size_t)s * Fo;
    float* po = out + (size_t)d * Fo;
    for (int f = lane; f < Fo; f += 32)
        atomicAdd(po + f, alpha * pl[f]);
}

// ---------------------------------------------------------------------------
// Optional LeakyReLU(0.01) + f32 -> f16 conversion (next layer's GEMM input)
// ---------------------------------------------------------------------------
__global__ void act_convert_kernel(const float* __restrict__ in, _Float16* __restrict__ out,
                                   int n, float slope, int act) {
    int i = blockIdx.x * blockDim.x + threadIdx.x;
    if (i >= n) return;
    float v = in[i];
    if (act && v < 0.f) v *= slope;
    out[i] = (_Float16)v;
}

// ---------------------------------------------------------------------------
// Per-graph max pool; batch = i/1250 exactly (uniform segments)
// ---------------------------------------------------------------------------
__global__ void pool_max_kernel(const float* __restrict__ h, float* __restrict__ g,
                                int npg, int F) {
    int graph = blockIdx.x;
    const float* base = h + (size_t)graph * npg * F;
    for (int f = threadIdx.x; f < F; f += blockDim.x) {
        float mx = -1e30f;
        for (int n = 0; n < npg; ++n) mx = fmaxf(mx, base[(size_t)n * F + f]);
        g[graph * F + f] = mx;
    }
}

// ---------------------------------------------------------------------------
// Tiny dense layer for the MLP head (M = 16 rows; scalar is optimal here)
// ---------------------------------------------------------------------------
__global__ void mlp_dense_kernel(const float* __restrict__ in, const float* __restrict__ W,
                                 const float* __restrict__ b, float* __restrict__ out,
                                 int M, int K, int Nc, int relu) {
    int idx = blockIdx.x * blockDim.x + threadIdx.x;
    if (idx >= M * Nc) return;
    int mr = idx / Nc, c = idx - mr * Nc;
    const float* a = in + (size_t)mr * K;
    float acc = b[c];
    for (int k = 0; k < K; ++k) acc += a[k] * W[(size_t)k * Nc + c];
    if (relu) acc = fmaxf(acc, 0.f);
    out[idx] = acc;
}

// ---------------------------------------------------------------------------
// Host orchestration
// Inputs (flattened pytree order): 0:x 1:edge_index 2:batch
//   3+4l..: (Wl, Wr, att, b) for l=0..5;   27..32: W1,b1,W2,b2,W3,b3
// ---------------------------------------------------------------------------
extern "C" void kernel_launch(void* const* d_in, const int* in_sizes, int n_in,
                              void* d_out, int out_size, void* d_ws, size_t ws_size,
                              hipStream_t stream) {
    (void)in_sizes; (void)n_in; (void)out_size; (void)ws_size;
    static const int DIMS[7] = {128, 32, 64, 128, 256, 512, 640};
    const int N = N_NODES, E = N_EDGES, G = N_GRAPHS, FMAX = 640;

    const float* x  = (const float*)d_in[0];
    const int* ei   = (const int*)d_in[1];
    const int* src  = ei;
    const int* dst  = ei + E;

    char* ws = (char*)d_ws;
    size_t off = 0;
    auto take = [&](size_t bytes) -> void* {
        void* p = ws + off;
        off = (off + bytes + 255) & ~(size_t)255;
        return p;
    };

    _Float16* hf16 = (_Float16*)take((size_t)N * FMAX * sizeof(_Float16));
    float* xl      = (float*)take((size_t)N * FMAX * sizeof(float));
    float* xr      = (float*)take((size_t)N * FMAX * sizeof(float));
    float* hout    = (float*)take((size_t)N * FMAX * sizeof(float));
    _Float16* wlf  = (_Float16*)take((size_t)512 * 640 * sizeof(_Float16));
    _Float16* wrf  = (_Float16*)take((size_t)512 * 640 * sizeof(_Float16));
    float* evals   = (float*)take((size_t)E * sizeof(float));
    float* exv     = (float*)take((size_t)E * sizeof(float));
    float* mmax    = (float*)take((size_t)N * sizeof(float));
    float* denom   = (float*)take((size_t)N * sizeof(float));
    float* g       = (float*)take((size_t)G * FMAX * sizeof(float));
    float* t1      = (float*)take((size_t)G * 1024 * sizeof(float));
    float* t2      = (float*)take((size_t)G * 512 * sizeof(float));

    auto cdiv = [](int a, int b) { return (a + b - 1) / b; };

    // x (f32) -> hf16, no activation
    act_convert_kernel<<<cdiv(N * 128, 256), 256, 0, stream>>>(x, hf16, N * 128, 0.01f, 0);

    for (int l = 0; l < 6; ++l) {
        int Fi = DIMS[l], Fo = DIMS[l + 1];
        const float* Wl  = (const float*)d_in[3 + 4 * l + 0];
        const float* Wr  = (const float*)d_in[3 + 4 * l + 1];
        const float* att = (const float*)d_in[3 + 4 * l + 2];
        const float* b   = (const float*)d_in[3 + 4 * l + 3];

        repack_w_kernel<<<cdiv(Fi * Fo, 256), 256, 0, stream>>>(Wl, wlf, Fi, Fo);
        repack_w_kernel<<<cdiv(Fi * Fo, 256), 256, 0, stream>>>(Wr, wrf, Fi, Fo);

        int mtiles = N / 16;
        int ntiles = Fo / 16;
        if (Fo == 32) {
            int waves = mtiles * (ntiles / 2);
            gemm_wmma_kernel<2><<<cdiv(waves * 32, 256), 256, 0, stream>>>(hf16, wlf, xl, N, Fi, Fo);
            gemm_wmma_kernel<2><<<cdiv(waves * 32, 256), 256, 0, stream>>>(hf16, wrf, xr, N, Fi, Fo);
        } else {
            int waves = mtiles * (ntiles / 4);
            gemm_wmma_kernel<4><<<cdiv(waves * 32, 256), 256, 0, stream>>>(hf16, wlf, xl, N, Fi, Fo);
            gemm_wmma_kernel<4><<<cdiv(waves * 32, 256), 256, 0, stream>>>(hf16, wrf, xr, N, Fi, Fo);
        }

        init_softmax_kernel<<<cdiv(N, 256), 256, 0, stream>>>(mmax, denom, N);
        init_bias_rows_kernel<<<cdiv(N * Fo, 256), 256, 0, stream>>>(hout, b, N, Fo);

        edge_scores_kernel<<<cdiv(E * 32, 256), 256, 0, stream>>>(xl, xr, src, dst, att,
                                                                  evals, mmax, E, Fo);
        edge_exp_kernel<<<cdiv(E, 256), 256, 0, stream>>>(evals, dst, mmax, exv, denom, E);
        edge_aggregate_kernel<<<cdiv(E * 32, 256), 256, 0, stream>>>(xl, exv, denom, src, dst,
                                                                     hout, E, Fo);
        if (l < 5)
            act_convert_kernel<<<cdiv(N * Fo, 256), 256, 0, stream>>>(hout, hf16, N * Fo, 0.01f, 1);
    }

    pool_max_kernel<<<G, 256, 0, stream>>>(hout, g, N / G, 640);

    const float* W1 = (const float*)d_in[27]; const float* b1 = (const float*)d_in[28];
    const float* W2 = (const float*)d_in[29]; const float* b2 = (const float*)d_in[30];
    const float* W3 = (const float*)d_in[31]; const float* b3 = (const float*)d_in[32];
    mlp_dense_kernel<<<cdiv(G * 1024, 256), 256, 0, stream>>>(g,  W1, b1, t1, G, 640, 1024, 1);
    mlp_dense_kernel<<<cdiv(G * 512, 256), 256, 0, stream>>>(t1, W2, b2, t2, G, 1024, 512, 1);
    mlp_dense_kernel<<<1, 64, 0, stream>>>(t2, W3, b3, (float*)d_out, G, 512, 4, 0);
}
